// Attention_54924041781472
// MI455X (gfx1250) — matrix operand order repaired and checked by reading, not compiled, not run on hardware
//
#include <hip/hip_runtime.h>

typedef __attribute__((ext_vector_type(16))) _Float16 v16h;
typedef __attribute__((ext_vector_type(8)))  _Float16 v8h;
typedef __attribute__((ext_vector_type(8)))  float    v8f;

#define WMMA_F16(A, B, C) \
  __builtin_amdgcn_wmma_f32_16x16x32_f16(false, (A), false, (B), (short)0, (C), false, false)

static __device__ __forceinline__ v16h cat8(v8h lo, v8h hi) {
  return __builtin_shufflevector(lo, hi, 0, 1, 2, 3, 4, 5, 6, 7, 8, 9, 10, 11, 12, 13, 14, 15);
}

// ---------------------------------------------------------------------------
// Weight convert + transpose: Wt[n,k] = (f16) W[k,n], 1024x1024.
// Makes all WMMA B-fragments a single contiguous 32B load per lane.
// ---------------------------------------------------------------------------
__global__ __launch_bounds__(256) void wconv_t(const float* __restrict__ W,
                                               _Float16* __restrict__ Wt) {
  const int idx = blockIdx.x * 256 + threadIdx.x;   // 0 .. 1M-1
  const int n = idx >> 10;
  const int k = idx & 1023;
  Wt[(size_t)n * 1024 + k] = (_Float16)W[(size_t)k * 1024 + n];
}

// ---------------------------------------------------------------------------
// Projection GEMM: Out = X(fp32 [4096,1024]) @ W (via Wt f16 [N,K]).
// vout==0: Out is f16 [B,H,T,64] (Q,K).  vout==1: Out is f16 [B,H,64,T] (V^T).
// One wave computes 4 N-tiles from a single A-fragment (4x A reuse):
// per k-step: 4 fp32 A-loads + 4 B-loads feed 4 WMMAs (~2 loads/WMMA).
// ---------------------------------------------------------------------------
__global__ __launch_bounds__(256) void proj_gemm(const float* __restrict__ X,
                                                 const _Float16* __restrict__ Wt,
                                                 _Float16* __restrict__ Out,
                                                 int vout) {
  const int lane   = threadIdx.x & 31;
  const int wv     = threadIdx.x >> 5;
  const int hl     = lane >> 4;     // half-wave select
  const int l16    = lane & 15;
  const int ntile0 = (blockIdx.x * 8 + wv) * 4;   // 4 tiles: ntile0..ntile0+3
  const int mtile  = blockIdx.y;                  // 0..255

  const float*    arow = X  + (size_t)(mtile * 16 + l16) * 1024;
  const _Float16* b0   = Wt + (size_t)((ntile0 + 0) * 16 + l16) * 1024 + hl * 16;
  const _Float16* b1   = Wt + (size_t)((ntile0 + 1) * 16 + l16) * 1024 + hl * 16;
  const _Float16* b2   = Wt + (size_t)((ntile0 + 2) * 16 + l16) * 1024 + hl * 16;
  const _Float16* b3   = Wt + (size_t)((ntile0 + 3) * 16 + l16) * 1024 + hl * 16;

  v8f acc0 = {}, acc1 = {}, acc2 = {}, acc3 = {};
  for (int k0 = 0; k0 < 1024; k0 += 32) {
    // WGP-scope prefetch of the next A chunk (locality=3 -> near caches).
    __builtin_prefetch(arow + k0 + 64, 0, 3);
    const float4 f0 = *(const float4*)(arow + k0 + hl * 16);
    const float4 f1 = *(const float4*)(arow + k0 + hl * 16 + 4);
    const float4 f2 = *(const float4*)(arow + k0 + hl * 16 + 8);
    const float4 f3 = *(const float4*)(arow + k0 + hl * 16 + 12);
    v16h a;
    a[0]  = (_Float16)f0.x; a[1]  = (_Float16)f0.y; a[2]  = (_Float16)f0.z; a[3]  = (_Float16)f0.w;
    a[4]  = (_Float16)f1.x; a[5]  = (_Float16)f1.y; a[6]  = (_Float16)f1.z; a[7]  = (_Float16)f1.w;
    a[8]  = (_Float16)f2.x; a[9]  = (_Float16)f2.y; a[10] = (_Float16)f2.z; a[11] = (_Float16)f2.w;
    a[12] = (_Float16)f3.x; a[13] = (_Float16)f3.y; a[14] = (_Float16)f3.z; a[15] = (_Float16)f3.w;
    acc0 = WMMA_F16(a, *(const v16h*)(b0 + k0), acc0);
    acc1 = WMMA_F16(a, *(const v16h*)(b1 + k0), acc1);
    acc2 = WMMA_F16(a, *(const v16h*)(b2 + k0), acc2);
    acc3 = WMMA_F16(a, *(const v16h*)(b3 + k0), acc3);
  }

#pragma unroll
  for (int tt = 0; tt < 4; ++tt) {
    const v8f acc = (tt == 0) ? acc0 : (tt == 1) ? acc1 : (tt == 2) ? acc2 : acc3;
    const int n = (ntile0 + tt) * 16 + l16;
    const int hcol = n >> 6, dcol = n & 63;
#pragma unroll
    for (int r = 0; r < 8; ++r) {
      const int mr = mtile * 16 + r + hl * 8;     // global row (b*T + t)
      const int bb = mr >> 11, t = mr & 2047;
      size_t off;
      if (vout) off = ((size_t)(bb * 16 + hcol) * 64 + dcol) * 2048 + t;    // [B,H,DV,T]
      else      off = ((size_t)(bb * 16 + hcol) * 2048 + t) * 64 + dcol;    // [B,H,T,DK]
      Out[off] = (_Float16)acc[r];
    }
  }
}

// ---------------------------------------------------------------------------
// Flash attention: one wave per (b, h, 16-row query block). 32 keys / step.
// Qp,Kp f16 [B,H,T,64]; Vt f16 [B,H,64,T]; Ob f16 [B,T,H*64] (merged heads).
// ---------------------------------------------------------------------------
__global__ __launch_bounds__(256) void attn_fa(const _Float16* __restrict__ Qp,
                                               const _Float16* __restrict__ Kp,
                                               const _Float16* __restrict__ Vt,
                                               _Float16* __restrict__ Ob) {
  __shared__ __align__(16) _Float16 lds[8 * 512];   // 16x32 f16 per wave
  const int lane = threadIdx.x & 31;
  const int wv   = threadIdx.x >> 5;
  const int hl   = lane >> 4;
  const int l16  = lane & 15;
  const int gw   = blockIdx.x * 8 + wv;          // 0..4095
  const int qblk = gw & 127;
  const int h    = (gw >> 7) & 15;
  const int bt   = gw >> 11;
  const size_t bh = (size_t)(bt * 16 + h);
  _Float16* pl = lds + wv * 512;

  // Q fragments (16 rows x 64 features) held in registers for the whole wave.
  const _Float16* qrow = Qp + (bh * 2048 + qblk * 16 + l16) * 64;
  const v16h qa0 = cat8(*(const v8h*)(qrow + hl * 16),      *(const v8h*)(qrow + 8 + hl * 16));
  const v16h qa1 = cat8(*(const v8h*)(qrow + 32 + hl * 16), *(const v8h*)(qrow + 40 + hl * 16));

  float rmax[8], rsum[8];
  v8f o0 = {}, o1 = {}, o2 = {}, o3 = {};
#pragma unroll
  for (int r = 0; r < 8; ++r) { rmax[r] = -1e30f; rsum[r] = 0.f; }

  const int nkeys = qblk * 16 + 16;               // causal extent of this block
  for (int j0 = 0; j0 < nkeys; j0 += 32) {
    // S = Q @ K^T for 32 keys (two 16-key tiles), K-dim 64 split in two.
    v8f s0 = {}, s1 = {};
    const _Float16* kr0 = Kp + (bh * 2048 + j0 + l16) * 64;
    const _Float16* kr1 = kr0 + 16 * 64;
    s0 = WMMA_F16(qa0, *(const v16h*)(kr0 + hl * 16), s0);
    s0 = WMMA_F16(qa1, *(const v16h*)(kr0 + 32 + hl * 16), s0);
    s1 = WMMA_F16(qa0, *(const v16h*)(kr1 + hl * 16), s1);
    s1 = WMMA_F16(qa1, *(const v16h*)(kr1 + 32 + hl * 16), s1);

    // Online softmax. Row m lives at element r across the 16 lanes of a half.
#pragma unroll
    for (int r = 0; r < 8; ++r) {
      const int mq  = qblk * 16 + r + hl * 8;
      const int tk0 = j0 + l16;
      const int tk1 = j0 + 16 + l16;
      float e0 = (tk0 <= mq) ? s0[r] * 0.125f : -1e30f;   // 1/sqrt(64)
      float e1 = (tk1 <= mq) ? s1[r] * 0.125f : -1e30f;
      float bm = fmaxf(e0, e1);
      bm = fmaxf(bm, __shfl_xor(bm, 1, 32));
      bm = fmaxf(bm, __shfl_xor(bm, 2, 32));
      bm = fmaxf(bm, __shfl_xor(bm, 4, 32));
      bm = fmaxf(bm, __shfl_xor(bm, 8, 32));
      const float nm = fmaxf(rmax[r], bm);
      const float sc = __expf(rmax[r] - nm);
      rmax[r] = nm;
      const float p0 = __expf(e0 - nm);
      const float p1 = __expf(e1 - nm);
      // stage P in wave-private LDS for C-layout -> A-layout re-shape
      pl[(r + hl * 8) * 32 + l16]      = (_Float16)p0;
      pl[(r + hl * 8) * 32 + 16 + l16] = (_Float16)p1;
      float ps = p0 + p1;
      ps += __shfl_xor(ps, 1, 32);
      ps += __shfl_xor(ps, 2, 32);
      ps += __shfl_xor(ps, 4, 32);
      ps += __shfl_xor(ps, 8, 32);
      rsum[r] = rsum[r] * sc + ps;
      o0[r] *= sc; o1[r] *= sc; o2[r] *= sc; o3[r] *= sc;
    }

    // Reload P as a 16x32 A-fragment (same-wave LDS ops are in order).
    const v16h pa = cat8(*(const v8h*)(pl + l16 * 32 + hl * 16),
                         *(const v8h*)(pl + l16 * 32 + 8 + hl * 16));

    // O += P @ V : 4 output tiles of 16 features; V^T rows are contiguous in T.
    const _Float16* vr = Vt + (bh * 64 + l16) * 2048 + j0 + hl * 16;
    o0 = WMMA_F16(pa, *(const v16h*)(vr), o0);
    o1 = WMMA_F16(pa, *(const v16h*)(vr + 16 * 2048), o1);
    o2 = WMMA_F16(pa, *(const v16h*)(vr + 32 * 2048), o2);
    o3 = WMMA_F16(pa, *(const v16h*)(vr + 48 * 2048), o3);
  }

#pragma unroll
  for (int r = 0; r < 8; ++r) {
    const float inv = 1.f / rsum[r];
    const int tq = qblk * 16 + r + hl * 8;
    _Float16* orow = Ob + ((size_t)bt * 2048 + tq) * 1024 + h * 64;
    orow[l16]      = (_Float16)(o0[r] * inv);
    orow[16 + l16] = (_Float16)(o1[r] * inv);
    orow[32 + l16] = (_Float16)(o2[r] * inv);
    orow[48 + l16] = (_Float16)(o3[r] * inv);
  }
}

// ---------------------------------------------------------------------------
// Output projection: d_out(fp32 [4096,1024]) = Ob(f16) @ Wo (via WoT f16 [N,K]).
// Same 4x N-tile register blocking as proj_gemm.
// ---------------------------------------------------------------------------
__global__ __launch_bounds__(256) void out_gemm(const _Float16* __restrict__ A,
                                                const _Float16* __restrict__ Wt,
                                                float* __restrict__ Out) {
  const int lane   = threadIdx.x & 31;
  const int wv     = threadIdx.x >> 5;
  const int hl     = lane >> 4;
  const int l16    = lane & 15;
  const int ntile0 = (blockIdx.x * 8 + wv) * 4;
  const int mtile  = blockIdx.y;

  const _Float16* arow = A  + (size_t)(mtile * 16 + l16) * 1024;
  const _Float16* b0   = Wt + (size_t)((ntile0 + 0) * 16 + l16) * 1024 + hl * 16;
  const _Float16* b1   = Wt + (size_t)((ntile0 + 1) * 16 + l16) * 1024 + hl * 16;
  const _Float16* b2   = Wt + (size_t)((ntile0 + 2) * 16 + l16) * 1024 + hl * 16;
  const _Float16* b3   = Wt + (size_t)((ntile0 + 3) * 16 + l16) * 1024 + hl * 16;

  v8f acc0 = {}, acc1 = {}, acc2 = {}, acc3 = {};
  for (int k0 = 0; k0 < 1024; k0 += 32) {
    __builtin_prefetch(arow + k0 + 64, 0, 3);
    const v16h a = cat8(*(const v8h*)(arow + k0 + hl * 16),
                        *(const v8h*)(arow + k0 + 8 + hl * 16));
    acc0 = WMMA_F16(a, *(const v16h*)(b0 + k0), acc0);
    acc1 = WMMA_F16(a, *(const v16h*)(b1 + k0), acc1);
    acc2 = WMMA_F16(a, *(const v16h*)(b2 + k0), acc2);
    acc3 = WMMA_F16(a, *(const v16h*)(b3 + k0), acc3);
  }

#pragma unroll
  for (int tt = 0; tt < 4; ++tt) {
    const v8f acc = (tt == 0) ? acc0 : (tt == 1) ? acc1 : (tt == 2) ? acc2 : acc3;
    const int n = (ntile0 + tt) * 16 + l16;
#pragma unroll
    for (int r = 0; r < 8; ++r) {
      const int mr = mtile * 16 + r + hl * 8;
      Out[(size_t)mr * 1024 + n] = acc[r];
    }
  }
}

// ---------------------------------------------------------------------------
extern "C" void kernel_launch(void* const* d_in, const int* in_sizes, int n_in,
                              void* d_out, int out_size, void* d_ws, size_t ws_size,
                              hipStream_t stream) {
  const float* q  = (const float*)d_in[0];
  const float* k  = (const float*)d_in[1];
  const float* v  = (const float*)d_in[2];
  // d_in[3] = causal mask, computed analytically in attn_fa
  const float* Wq = (const float*)d_in[4];
  const float* Wk = (const float*)d_in[5];
  const float* Wv = (const float*)d_in[6];
  const float* Wo = (const float*)d_in[7];

  char* ws = (char*)d_ws;
  const size_t MB = 1024 * 1024;
  _Float16* WqT = (_Float16*)(ws + 0 * MB);    // 2 MB each
  _Float16* WkT = (_Float16*)(ws + 2 * MB);
  _Float16* WvT = (_Float16*)(ws + 4 * MB);
  _Float16* WoT = (_Float16*)(ws + 6 * MB);
  _Float16* Qp  = (_Float16*)(ws + 8 * MB);    // 8 MB each
  _Float16* Kp  = (_Float16*)(ws + 16 * MB);
  _Float16* Vtr = (_Float16*)(ws + 24 * MB);
  _Float16* Ob  = (_Float16*)(ws + 32 * MB);   // total 40 MB

  wconv_t<<<4096, 256, 0, stream>>>(Wq, WqT);
  wconv_t<<<4096, 256, 0, stream>>>(Wk, WkT);
  wconv_t<<<4096, 256, 0, stream>>>(Wv, WvT);
  wconv_t<<<4096, 256, 0, stream>>>(Wo, WoT);

  // 64 N-tiles / (8 waves * 4 tiles) = 2 blocks in x; 256 M-tiles in y.
  proj_gemm<<<dim3(2, 256), 256, 0, stream>>>(q, WqT, Qp, 0);
  proj_gemm<<<dim3(2, 256), 256, 0, stream>>>(k, WkT, Kp, 0);
  proj_gemm<<<dim3(2, 256), 256, 0, stream>>>(v, WvT, Vtr, 1);

  attn_fa<<<512, 256, 0, stream>>>(Qp, Kp, Vtr, Ob);

  out_gemm<<<dim3(2, 256), 256, 0, stream>>>(Ob, WoT, (float*)d_out);
}
